// VectorQuantizer_30872224923678
// MI455X (gfx1250) — matrix operand order repaired and checked by reading, not compiled, not run on hardware
//
#include <hip/hip_runtime.h>

typedef __attribute__((ext_vector_type(16))) __bf16 v16bf;
typedef __attribute__((ext_vector_type(8)))  float  v8f;

#define NUM_E   1024
#define DIM     64
#define HW      4096            // 64*64
#define CHW     (64*4096)
#define N_TOTAL 65536
#define ROWS_PER_WG 128
#define KSTAGE  128
#define NSTAGES (NUM_E / KSTAGE)

static __device__ __forceinline__ v8f wmma_bf16(v16bf a, v16bf b, v8f c) {
  // (neg_a, A, neg_b, B, c_mod, C, reuse_a, reuse_b)
  return __builtin_amdgcn_wmma_f32_16x16x32_bf16(false, a, false, b, (short)0, c,
                                                 false, false);
}

union FragU { __bf16 e[16]; v16bf v; };

__global__ void vq_zero_loss(float* lossp) { *lossp = 0.0f; }

__global__ __launch_bounds__(256)
void vq_main(const float* __restrict__ z, const float* __restrict__ emb,
             float* __restrict__ qout, float* __restrict__ lossp,
             float* __restrict__ idxout)
{
  __shared__ __align__(32) __bf16 sHi[KSTAGE * DIM];   // 16 KB
  __shared__ __align__(32) __bf16 sLo[KSTAGE * DIM];   // 16 KB
  __shared__ float sNorm[KSTAGE];                      // 512 B
  __shared__ int   sKstar[128];                        // 512 B

  const int tid  = threadIdx.x;
  const int lane = tid & 31;
  const int wave = tid >> 5;
  const int mrow = lane & 15;
  const int rowbase = blockIdx.x * ROWS_PER_WG + wave * 16;

  // ---- Build per-wave A fragments from z (NCHW -> row-major 16x64 tile),
  //      error-compensated bf16 split: x = hi + lo. ----
  const int n  = rowbase + mrow;
  const int zb = n >> 12;           // batch index (HW = 4096)
  const int hw = n & (HW - 1);
  const float* zp = z + (size_t)zb * CHW + hw;
  const int cbase = (lane < 16) ? 0 : 16;  // contiguous k layout (matches B)

  FragU ah0, ah1, al0, al1;
  #pragma unroll
  for (int j = 0; j < 8; ++j) {
    float x0 = zp[(size_t)(cbase + j)      * HW];
    float x1 = zp[(size_t)(cbase + 8 + j)  * HW];
    float x2 = zp[(size_t)(cbase + 32 + j) * HW];
    float x3 = zp[(size_t)(cbase + 40 + j) * HW];
    __bf16 h;
    h = (__bf16)x0; ah0.e[j]     = h; al0.e[j]     = (__bf16)(x0 - (float)h);
    h = (__bf16)x1; ah0.e[8 + j] = h; al0.e[8 + j] = (__bf16)(x1 - (float)h);
    h = (__bf16)x2; ah1.e[j]     = h; al1.e[j]     = (__bf16)(x2 - (float)h);
    h = (__bf16)x3; ah1.e[8 + j] = h; al1.e[8 + j] = (__bf16)(x3 - (float)h);
  }

  float minv[8]; int mini[8];
  #pragma unroll
  for (int i = 0; i < 8; ++i) { minv[i] = 3.4e38f; mini[i] = 0; }

  const int bcb = (lane < 16) ? 0 : 16;    // contiguous k layout c-offset (matches A)

  for (int stage = 0; stage < NSTAGES; ++stage) {
    __syncthreads();
    {   // cooperative stage: 128 codebook rows -> LDS bf16 hi/lo + ||e||^2
      const int kl = tid >> 1;
      const int ch = (tid & 1) * 32;
      const float* ep = emb + (size_t)(stage * KSTAGE + kl) * DIM + ch;
      float ss = 0.0f;
      #pragma unroll
      for (int j = 0; j < 32; ++j) {
        float x = ep[j];
        __bf16 h = (__bf16)x;
        sHi[kl * DIM + ch + j] = h;
        sLo[kl * DIM + ch + j] = (__bf16)(x - (float)h);
        ss = fmaf(x, x, ss);
      }
      ss += __shfl_xor(ss, 1, 32);
      if ((tid & 1) == 0) sNorm[kl] = ss;
    }
    __syncthreads();

    #pragma unroll 2
    for (int t = 0; t < KSTAGE / 16; ++t) {
      const int kl = t * 16 + mrow;
      const float ensq = sNorm[kl];
      const __bf16* hrow = &sHi[kl * DIM + bcb];
      const __bf16* lrow = &sLo[kl * DIM + bcb];
      v16bf bh0 = *(const v16bf*)(hrow);
      v16bf bh1 = *(const v16bf*)(hrow + 32);
      v16bf bl0 = *(const v16bf*)(lrow);
      v16bf bl1 = *(const v16bf*)(lrow + 32);
      v8f acc = {};
      acc = wmma_bf16(ah0.v, bh0, acc);   // hi*hi, c 0..31
      acc = wmma_bf16(ah1.v, bh1, acc);   // hi*hi, c 32..63
      acc = wmma_bf16(ah0.v, bl0, acc);   // hi*lo
      acc = wmma_bf16(ah1.v, bl1, acc);
      acc = wmma_bf16(al0.v, bh0, acc);   // lo*hi
      acc = wmma_bf16(al1.v, bh1, acc);
      const int kg = stage * KSTAGE + kl;
      #pragma unroll
      for (int i = 0; i < 8; ++i) {
        float d = fmaf(-2.0f, acc[i], ensq);   // ||e||^2 - 2 z.e
        if (d < minv[i]) { minv[i] = d; mini[i] = kg; }
      }
    }
  }

  // ---- argmin reduction across the 16 lanes that share a row ----
  #pragma unroll
  for (int i = 0; i < 8; ++i) {
    float v = minv[i]; int id = mini[i];
    #pragma unroll
    for (int ms = 0; ms < 4; ++ms) {
      const int mask = 1 << ms;                 // stays within 16-lane half
      float ov = __shfl_xor(v, mask, 32);
      int   oi = __shfl_xor(id, mask, 32);
      if (ov < v || (ov == v && oi < id)) { v = ov; id = oi; }
    }
    minv[i] = v; mini[i] = id;
  }

  if (lane == 0) {        // rows 0..7 live in half 0
    #pragma unroll
    for (int i = 0; i < 8; ++i) sKstar[wave * 16 + i] = mini[i];
  }
  if (lane == 16) {       // rows 8..15 live in half 1
    #pragma unroll
    for (int i = 0; i < 8; ++i) sKstar[wave * 16 + 8 + i] = mini[i];
  }
  __syncthreads();

  if (lane < 16) idxout[rowbase + lane] = (float)sKstar[wave * 16 + lane];

  // ---- gather E[k*], write quantized (NCHW, coalesced over w), loss ----
  const int   kst  = sKstar[wave * 16 + mrow];
  const float* erow = emb + (size_t)kst * DIM;
  const int   codd = lane >> 4;          // lanes 0-15: even c, 16-31: odd c
  float s = 0.0f;
  #pragma unroll 8
  for (int cc = 0; cc < 32; ++cc) {
    const int c = 2 * cc + codd;
    const float q = erow[c];
    const size_t o = (size_t)zb * CHW + (size_t)c * HW + hw;
    qout[o] = q;                          // straight-through output == q
    const float dz = q - z[o];
    s = fmaf(dz, dz, s);
  }
  s += __shfl_xor(s, 1, 32);
  s += __shfl_xor(s, 2, 32);
  s += __shfl_xor(s, 4, 32);
  s += __shfl_xor(s, 8, 32);
  s += __shfl_xor(s, 16, 32);
  if (lane == 0) atomicAdd(lossp, s * (1.25f / 4194304.0f));
}

extern "C" void kernel_launch(void* const* d_in, const int* in_sizes, int n_in,
                              void* d_out, int out_size, void* d_ws, size_t ws_size,
                              hipStream_t stream) {
  const float* z   = (const float*)d_in[0];   // [16,64,64,64] f32
  const float* emb = (const float*)d_in[1];   // [1024,64] f32
  float* qout   = (float*)d_out;              // 4,194,304 floats (NCHW)
  float* lossp  = qout + (size_t)16 * 64 * 64 * 64;
  float* idxout = lossp + 1;                  // 65,536 floats

  vq_zero_loss<<<1, 1, 0, stream>>>(lossp);
  vq_main<<<N_TOTAL / ROWS_PER_WG, 256, 0, stream>>>(z, emb, qout, lossp, idxout);
}
